// CrossAttention_44117904064564
// MI455X (gfx1250) — compile-verified, matrix operand order checked by reading, not run on hardware
//
#include <hip/hip_runtime.h>

typedef __attribute__((ext_vector_type(16))) _Float16 v16h;
typedef __attribute__((ext_vector_type(8)))  _Float16 v8h;
typedef __attribute__((ext_vector_type(8)))  float    v8f;
typedef __attribute__((ext_vector_type(4)))  float    v4f;

#define WMMA_F16(a, b, c) \
  __builtin_amdgcn_wmma_f32_16x16x32_f16(false, (a), false, (b), (short)0, (c), false, false)

// Load one 16x32 f16 A/B fragment (row-major storage, row pointer already
// offset by half*8 halves). Matches ISA 16-bit A-matrix layout:
//   lane&15 = row, elems 0..7 = K (hf*8 ..), elems 8..15 = K+16.
__device__ __forceinline__ v16h load_frag(const _Float16* p) {
  v8h lo = *(const v8h*)(p);
  v8h hi = *(const v8h*)(p + 16);
  v16h f;
#pragma unroll
  for (int i = 0; i < 8; ++i) { f[i] = lo[i]; f[i + 8] = hi[i]; }
  return f;
}

// ---------------------------------------------------------------------------
// Vectorized f32 -> f16: 8 elements per thread (2x b128 load, 1x b128 store)
__global__ __launch_bounds__(256) void cvt_f32_f16_v8(const float* __restrict__ in,
                                                      _Float16* __restrict__ out, int n8) {
  int i = blockIdx.x * 256 + threadIdx.x;
  if (i < n8) {
    v4f a = ((const v4f*)in)[2 * i];
    v4f b = ((const v4f*)in)[2 * i + 1];
    v8h o;
#pragma unroll
    for (int j = 0; j < 4; ++j) { o[j] = (_Float16)a[j]; o[4 + j] = (_Float16)b[j]; }
    ((v8h*)out)[i] = o;
  }
}

// LDS-tiled coalesced transpose: Wt[n*K + k] = (f16) W[k*N + n]
// grid = (N/32, K/32), block = (32, 8)
__global__ __launch_bounds__(256) void transpose_cvt(const float* __restrict__ W,
                                                     _Float16* __restrict__ Wt,
                                                     int K, int N) {
  __shared__ float tile[32][33];
  const int tx = threadIdx.x, ty = threadIdx.y;
  const int n0 = blockIdx.x * 32, k0 = blockIdx.y * 32;
#pragma unroll
  for (int t = 0; t < 4; ++t)
    tile[ty + t * 8][tx] = W[(size_t)(k0 + ty + t * 8) * N + n0 + tx];
  __syncthreads();
#pragma unroll
  for (int t = 0; t < 4; ++t)
    Wt[(size_t)(n0 + ty + t * 8) * K + k0 + tx] = (_Float16)tile[tx][ty + t * 8];
}

// ---------------------------------------------------------------------------
// GEMM: C[M,N] = A[M,K] * Bt[N,K]^T + bias[N], 64x64 tile per wave.
// Per K=32 step: 8 fragment loads (8x2 b128) feed 16 WMMAs (1 b128/WMMA).
// MODE 0: f16 out [M,N]   MODE 1: f16 out as Vt[b][h][d][nk]   MODE 2: f32 out
template <int MODE>
__global__ __launch_bounds__(256) void gemm_wmma(
    const _Float16* __restrict__ A, const _Float16* __restrict__ Bt,
    const float* __restrict__ bias, _Float16* __restrict__ Ch,
    float* __restrict__ Cf, int M, int N, int K) {
  const int lane = threadIdx.x & 31;
  const int wave = threadIdx.x >> 5;
  const int gw = blockIdx.x * (blockDim.x >> 5) + wave;
  const int tilesN = N >> 6;
  const int tm = gw / tilesN;
  const int tn = gw - tm * tilesN;
  const int m0 = tm << 6, n0 = tn << 6;
  if (m0 >= M) return;
  const int r = lane & 15, hf = lane >> 4;

  const _Float16* aP[4];
  const _Float16* bP[4];
#pragma unroll
  for (int j = 0; j < 4; ++j) {
    aP[j] = A + (size_t)(m0 + j * 16 + r) * K + hf * 8;
    bP[j] = Bt + (size_t)(n0 + j * 16 + r) * K + hf * 8;
  }

  v8f acc[4][4];
#pragma unroll
  for (int j = 0; j < 4; ++j)
#pragma unroll
    for (int l = 0; l < 4; ++l) acc[j][l] = (v8f){};

  for (int k = 0; k < K; k += 32) {
    v16h a[4], b[4];
#pragma unroll
    for (int j = 0; j < 4; ++j) a[j] = load_frag(aP[j] + k);
#pragma unroll
    for (int l = 0; l < 4; ++l) b[l] = load_frag(bP[l] + k);
#pragma unroll
    for (int j = 0; j < 4; ++j)
#pragma unroll
      for (int l = 0; l < 4; ++l) acc[j][l] = WMMA_F16(a[j], b[l], acc[j][l]);
  }

  float bv[4];
#pragma unroll
  for (int l = 0; l < 4; ++l) bv[l] = bias[n0 + l * 16 + r];

#pragma unroll
  for (int j = 0; j < 4; ++j) {
#pragma unroll
    for (int l = 0; l < 4; ++l) {
#pragma unroll
      for (int i = 0; i < 8; ++i) {
        const int row = m0 + j * 16 + i + 8 * hf;
        const int col = n0 + l * 16 + r;
        const float v = acc[j][l][i] + bv[l];
        if (MODE == 0) {
          Ch[(size_t)row * N + col] = (_Float16)v;
        } else if (MODE == 2) {
          Cf[(size_t)row * N + col] = v;
        } else {
          // V output transposed per head: Vt[((b*16 + h)*64 + d)*1024 + nk]
          const int bb = row >> 10, nk = row & 1023;
          const int hd = col >> 6, d = col & 63;
          Ch[(size_t)((bb * 16 + hd) * 64 + d) * 1024 + nk] = (_Float16)v;
        }
      }
    }
  }
}

// ---------------------------------------------------------------------------
// Flash attention: one wave per (b, h, 16-row query tile); key blocks of 64.
// Q,K: [B*N, 1024] f16 (head cols h*64..h*64+63); Vt: [B,H,64,Nk] f16.
// Softmax runs in exp2 domain (SCALE*log2e folded into one multiply).
// Per 64-key block: 16 WMMAs, one pair of shfl reduction trees, one O-rescale.
__global__ __launch_bounds__(256) void flash_attn(
    const _Float16* __restrict__ Q, const _Float16* __restrict__ Km,
    const _Float16* __restrict__ Vt, _Float16* __restrict__ O) {
  constexpr int Nq = 4096, Nk = 1024, H = 16, HD = 64, LD = 1024;
  constexpr float C2 = 0.125f * 1.44269504088896340736f;  // 64^-0.5 * log2(e)
  __shared__ _Float16 pbuf[8][16 * 64];

  const int lane = threadIdx.x & 31;
  const int wave = threadIdx.x >> 5;
  const int gw = blockIdx.x * 8 + wave;
  const int qt = gw & ((Nq / 16) - 1);  // 256 q-tiles per (b,h)
  const int bh = gw >> 8;
  const int hh = bh & (H - 1);
  const int b = bh >> 4;
  const int r = lane & 15, hf = lane >> 4;

  const _Float16* qp = Q + (size_t)(b * Nq + qt * 16 + r) * LD + hh * HD + hf * 8;
  const v16h aq0 = load_frag(qp);       // dims 0..31
  const v16h aq1 = load_frag(qp + 32);  // dims 32..63

  const _Float16* kbase = Km + (size_t)(b * Nk) * LD + hh * HD + hf * 8;
  const _Float16* vbase = Vt + (size_t)((b * H + hh) * HD + r) * Nk + hf * 8;

  v8f o[4];
#pragma unroll
  for (int t = 0; t < 4; ++t) o[t] = (v8f){};
  float mrow[8], lrow[8];
#pragma unroll
  for (int i = 0; i < 8; ++i) { mrow[i] = -1e30f; lrow[i] = 0.f; }

  _Float16* myp = &pbuf[wave][0];

  for (int kb = 0; kb < Nk; kb += 64) {
    // ---- S = Q * K^T for 64 keys (4 column tiles, K stored row-major)
    v8f s[4];
#pragma unroll
    for (int c = 0; c < 4; ++c) {
      const _Float16* kp = kbase + (size_t)(kb + c * 16 + r) * LD;
      const v16h ka = load_frag(kp);
      const v16h kbb = load_frag(kp + 32);
      v8f t = {};
      t = WMMA_F16(aq0, ka, t);
      t = WMMA_F16(aq1, kbb, t);
      s[c] = t;
    }

    // ---- online softmax stats (rows live across the 16-lane half-wave)
    float tmax[8];
#pragma unroll
    for (int i = 0; i < 8; ++i) {
      s[0][i] *= C2; s[1][i] *= C2; s[2][i] *= C2; s[3][i] *= C2;
      tmax[i] = fmaxf(fmaxf(s[0][i], s[1][i]), fmaxf(s[2][i], s[3][i]));
    }
#pragma unroll
    for (int off = 1; off < 16; off <<= 1)
#pragma unroll
      for (int i = 0; i < 8; ++i)
        tmax[i] = fmaxf(tmax[i], __shfl_xor(tmax[i], off, 32));

    float tsum[8];
#pragma unroll
    for (int i = 0; i < 8; ++i) {
      const float mn = fmaxf(mrow[i], tmax[i]);
      const float alpha = __builtin_amdgcn_exp2f(mrow[i] - mn);
      mrow[i] = mn;
      s[0][i] = __builtin_amdgcn_exp2f(s[0][i] - mn);
      s[1][i] = __builtin_amdgcn_exp2f(s[1][i] - mn);
      s[2][i] = __builtin_amdgcn_exp2f(s[2][i] - mn);
      s[3][i] = __builtin_amdgcn_exp2f(s[3][i] - mn);
      tsum[i] = (s[0][i] + s[1][i]) + (s[2][i] + s[3][i]);
      lrow[i] *= alpha;
      o[0][i] *= alpha; o[1][i] *= alpha; o[2][i] *= alpha; o[3][i] *= alpha;
    }
#pragma unroll
    for (int off = 1; off < 16; off <<= 1)
#pragma unroll
      for (int i = 0; i < 8; ++i)
        tsum[i] += __shfl_xor(tsum[i], off, 32);
#pragma unroll
    for (int i = 0; i < 8; ++i) lrow[i] += tsum[i];

    // ---- P: C-layout f32 -> row-major 16x64 f16 in LDS -> A-layout frags
#pragma unroll
    for (int i = 0; i < 8; ++i) {
      const int row = i + 8 * hf;
      myp[row * 64 + r]      = (_Float16)s[0][i];
      myp[row * 64 + 16 + r] = (_Float16)s[1][i];
      myp[row * 64 + 32 + r] = (_Float16)s[2][i];
      myp[row * 64 + 48 + r] = (_Float16)s[3][i];
    }
    asm volatile("s_wait_dscnt 0" ::: "memory");  // same-wave DS RAW order
    const v16h ap0 = load_frag(myp + r * 64 + hf * 8);       // keys 0..31
    const v16h ap1 = load_frag(myp + r * 64 + 32 + hf * 8);  // keys 32..63

    // ---- O += P * V  (Vt is transposed-B storage: rows=dims, cols=keys)
    const _Float16* vp = vbase + kb;
#pragma unroll
    for (int t = 0; t < 4; ++t) {
      const v16h bva = load_frag(vp + (size_t)(t * 16) * Nk);
      const v16h bvb = load_frag(vp + (size_t)(t * 16) * Nk + 32);
      o[t] = WMMA_F16(ap0, bva, o[t]);
      o[t] = WMMA_F16(ap1, bvb, o[t]);
    }
  }

  // ---- normalize + store [B*Nq, 1024] f16
  _Float16* obase = O + (size_t)(b * Nq + qt * 16) * LD + hh * HD;
#pragma unroll
  for (int i = 0; i < 8; ++i) {
    const float inv = 1.0f / lrow[i];
    const int row = i + 8 * hf;
    obase[(size_t)row * LD + 0 + r]  = (_Float16)(o[0][i] * inv);
    obase[(size_t)row * LD + 16 + r] = (_Float16)(o[1][i] * inv);
    obase[(size_t)row * LD + 32 + r] = (_Float16)(o[2][i] * inv);
    obase[(size_t)row * LD + 48 + r] = (_Float16)(o[3][i] * inv);
  }
}

// ---------------------------------------------------------------------------
extern "C" void kernel_launch(void* const* d_in, const int* in_sizes, int n_in,
                              void* d_out, int out_size, void* d_ws, size_t ws_size,
                              hipStream_t stream) {
  const float* query = (const float*)d_in[0];
  const float* key   = (const float*)d_in[1];
  const float* value = (const float*)d_in[2];
  const float* Wq = (const float*)d_in[3];
  const float* bq = (const float*)d_in[4];
  const float* Wk = (const float*)d_in[5];
  const float* bk = (const float*)d_in[6];
  const float* Wv = (const float*)d_in[7];
  const float* bv = (const float*)d_in[8];
  const float* Wo = (const float*)d_in[9];
  const float* bo = (const float*)d_in[10];
  float* out = (float*)d_out;

  char* ws = (char*)d_ws;
  _Float16* qh   = (_Float16*)(ws + 0);          // 16384x1024
  _Float16* kh   = (_Float16*)(ws + 33554432);   //  4096x768
  _Float16* vh   = (_Float16*)(ws + 39845888);   //  4096x768
  _Float16* WqT  = (_Float16*)(ws + 46137344);   //  1024x1024
  _Float16* WkT  = (_Float16*)(ws + 48234496);   //  1024x768
  _Float16* WvT  = (_Float16*)(ws + 49807360);   //  1024x768
  _Float16* WoT  = (_Float16*)(ws + 51380224);   //  1024x1024
  _Float16* Qp   = (_Float16*)(ws + 53477376);   // 16384x1024
  _Float16* Kp   = (_Float16*)(ws + 87031808);   //  4096x1024
  _Float16* Vtp  = (_Float16*)(ws + 95420416);   //  [4,16,64,1024]
  _Float16* Attn = (_Float16*)(ws + 103809024);  // 16384x1024

  // f32 -> f16 conversions (8 elems/thread)
  cvt_f32_f16_v8<<<2097152 / 256, 256, 0, stream>>>(query, qh, 2097152);
  cvt_f32_f16_v8<<<393216 / 256, 256, 0, stream>>>(key, kh, 393216);
  cvt_f32_f16_v8<<<393216 / 256, 256, 0, stream>>>(value, vh, 393216);

  // Coalesced LDS-tiled weight transposes
  transpose_cvt<<<dim3(32, 32), dim3(32, 8), 0, stream>>>(Wq, WqT, 1024, 1024);
  transpose_cvt<<<dim3(32, 24), dim3(32, 8), 0, stream>>>(Wk, WkT, 768, 1024);
  transpose_cvt<<<dim3(32, 24), dim3(32, 8), 0, stream>>>(Wv, WvT, 768, 1024);
  transpose_cvt<<<dim3(32, 32), dim3(32, 8), 0, stream>>>(Wo, WoT, 1024, 1024);

  // Projections (64x64 wave tiles, 8 waves/block)
  gemm_wmma<0><<<512, 256, 0, stream>>>(qh, WqT, bq, Qp, nullptr, 16384, 1024, 1024);
  gemm_wmma<0><<<128, 256, 0, stream>>>(kh, WkT, bk, Kp, nullptr, 4096, 1024, 768);
  gemm_wmma<1><<<128, 256, 0, stream>>>(vh, WvT, bv, Vtp, nullptr, 4096, 1024, 768);

  // Attention: 4*16*(4096/16) = 16384 waves -> 2048 blocks of 8 waves
  flash_attn<<<2048, 256, 0, stream>>>(Qp, Kp, Vtp, Attn);

  // Output projection -> f32 d_out
  gemm_wmma<2><<<512, 256, 0, stream>>>(Attn, WoT, bo, nullptr, out, 16384, 1024, 1024);
}
